// DegradationAttention_11965778886679
// MI455X (gfx1250) — compile-verified, matrix operand order checked by reading, not run on hardware
//
#include <hip/hip_runtime.h>
#include <hip/hip_bf16.h>

typedef __attribute__((ext_vector_type(16))) _Float16 v16h;
typedef __attribute__((ext_vector_type(8)))  _Float16 v8h;
typedef __attribute__((ext_vector_type(8)))  float    v8f;

union U16H { v16h v; v8h h8[2]; _Float16 h[16]; };
union U8H  { v8h  v; _Float16 h[8]; };
union U8F  { v8f  v; float    f[8]; };

#define L_SZ   2048
#define H_SZ   8
#define E_SZ   64
#define MTILE  128          // query rows per block (8 waves x 16)
#define STILE  32           // kv rows per step
#define NEG_INF_C (-4294967295.0f)   // -2^32 + 1
#define SCALEF 0.125f                // 1/sqrt(64)

__global__ __launch_bounds__(256)
void degradation_attn_fwd(const float* __restrict__ Q,
                          const float* __restrict__ K,
                          const float* __restrict__ V,
                          float* __restrict__ Out)
{
  __shared__ __align__(16) _Float16 sK [STILE * E_SZ];   // K tile, row-major f16
  __shared__ __align__(16) _Float16 sVT[E_SZ * STILE];   // V tile, transposed f16
  __shared__ __align__(16) _Float16 sP [8][16 * STILE];  // per-wave P tile (C->A relayout)
  __shared__ float sKN[STILE];                           // K row norms

  const int tid  = threadIdx.x;
  const int lane = tid & 31;
  const int wave = tid >> 5;
  const int r16  = lane & 15;     // column index inside 16-wide fragments
  const int hi   = lane >> 4;     // lane-group (k/row-half selector)

  const int n  = blockIdx.y;      // b*H + h, 0..15
  const int b  = n >> 3;
  const int h  = n & 7;
  const int l_block = blockIdx.x * MTILE;
  const int m0 = l_block + wave * 16;   // wave's first absolute query row

  // ------------ Q fragments (A layout, f32 -> f16) + row norms ------------
  const float* qrow = Q + ((size_t)(n * L_SZ + m0 + r16)) * E_SZ;

  U16H qfrag[2];
  #pragma unroll
  for (int ks = 0; ks < 2; ++ks) {
    #pragma unroll
    for (int ch = 0; ch < 2; ++ch) {
      const int e0 = ks * 32 + ch * 16 + hi * 8;
      float4 f0 = *(const float4*)(qrow + e0);
      float4 f1 = *(const float4*)(qrow + e0 + 4);
      const int o = ch * 8;
      qfrag[ks].h[o+0] = (_Float16)f0.x; qfrag[ks].h[o+1] = (_Float16)f0.y;
      qfrag[ks].h[o+2] = (_Float16)f0.z; qfrag[ks].h[o+3] = (_Float16)f0.w;
      qfrag[ks].h[o+4] = (_Float16)f1.x; qfrag[ks].h[o+5] = (_Float16)f1.y;
      qfrag[ks].h[o+6] = (_Float16)f1.z; qfrag[ks].h[o+7] = (_Float16)f1.w;
    }
  }

  float qn_own = 0.f;
  #pragma unroll
  for (int e = 0; e < E_SZ; e += 4) {
    float4 f = *(const float4*)(qrow + e);
    qn_own += f.x*f.x + f.y*f.y + f.z*f.z + f.w*f.w;
  }
  float qn8[8];
  #pragma unroll
  for (int j = 0; j < 8; ++j) qn8[j] = __shfl(qn_own, j + 8 * hi, 32);

  float m_i[8], l_i[8];
  v8f acc[4];
  #pragma unroll
  for (int j = 0; j < 8; ++j) { m_i[j] = -3.0e38f; l_i[j] = 0.f; }
  {
    v8f z = {};
    #pragma unroll
    for (int cc = 0; cc < 4; ++cc) acc[cc] = z;
  }

  const int s_end = l_block + MTILE;     // causal: block needs s < l_block+128

  for (int s_base = 0; s_base < s_end; s_base += STILE) {
    // ---------------- cooperative staging of K & V tiles ----------------
    {
      const int kr = tid >> 3;          // 0..31 : kv row within tile
      const int c8 = (tid & 7) << 3;    // 0..56 : 8-wide column chunk
      const float* krow = K + ((size_t)(n * L_SZ + s_base + kr)) * E_SZ + c8;
      float4 f0 = *(const float4*)(krow);
      float4 f1 = *(const float4*)(krow + 4);
      U8H kh;
      kh.h[0]=(_Float16)f0.x; kh.h[1]=(_Float16)f0.y; kh.h[2]=(_Float16)f0.z; kh.h[3]=(_Float16)f0.w;
      kh.h[4]=(_Float16)f1.x; kh.h[5]=(_Float16)f1.y; kh.h[6]=(_Float16)f1.z; kh.h[7]=(_Float16)f1.w;
      *(v8h*)&sK[kr * E_SZ + c8] = kh.v;

      float pn = f0.x*f0.x + f0.y*f0.y + f0.z*f0.z + f0.w*f0.w
               + f1.x*f1.x + f1.y*f1.y + f1.z*f1.z + f1.w*f1.w;
      pn += __shfl_xor(pn, 1, 32);
      pn += __shfl_xor(pn, 2, 32);
      pn += __shfl_xor(pn, 4, 32);
      if ((tid & 7) == 0) sKN[kr] = pn;

      const float* vrow = V + (((size_t)(b * L_SZ + s_base + kr)) * H_SZ + h) * E_SZ + c8;
      float4 g0 = *(const float4*)(vrow);
      float4 g1 = *(const float4*)(vrow + 4);
      sVT[(c8+0)*STILE + kr] = (_Float16)g0.x;
      sVT[(c8+1)*STILE + kr] = (_Float16)g0.y;
      sVT[(c8+2)*STILE + kr] = (_Float16)g0.z;
      sVT[(c8+3)*STILE + kr] = (_Float16)g0.w;
      sVT[(c8+4)*STILE + kr] = (_Float16)g1.x;
      sVT[(c8+5)*STILE + kr] = (_Float16)g1.y;
      sVT[(c8+6)*STILE + kr] = (_Float16)g1.z;
      sVT[(c8+7)*STILE + kr] = (_Float16)g1.w;

      if (s_base + STILE < s_end) {   // emits global_prefetch_b8 for next tile
        __builtin_prefetch(krow + STILE * E_SZ, 0, 1);
        __builtin_prefetch(vrow + (size_t)STILE * H_SZ * E_SZ, 0, 1);
      }
    }
    __syncthreads();

    if (s_base <= m0 + 15) {              // wave-uniform: EXEC stays all-ones
      const bool sub1 = (s_base + 16) <= (m0 + 15);

      float x0[8], x1[8];
      auto score_tile = [&](int c, float* xs) {
        // B fragment = K^T : lane holds K row (c+r16), e in A/B lane pattern
        const _Float16* kb = &sK[(c + r16) * E_SZ];
        U16H b0, b1;
        b0.h8[0] = *(const v8h*)(kb + 0  + hi*8);
        b0.h8[1] = *(const v8h*)(kb + 16 + hi*8);
        b1.h8[0] = *(const v8h*)(kb + 32 + hi*8);
        b1.h8[1] = *(const v8h*)(kb + 48 + hi*8);
        v8f cacc = {};
        cacc = __builtin_amdgcn_wmma_f32_16x16x32_f16(false, qfrag[0].v, false, b0.v,
                                                      (short)0, cacc, false, false);
        cacc = __builtin_amdgcn_wmma_f32_16x16x32_f16(false, qfrag[1].v, false, b1.v,
                                                      (short)0, cacc, false, false);
        const float kn   = sKN[c + r16];
        const int s_abs  = s_base + c + r16;
        U8F ca; ca.v = cacc;
        #pragma unroll
        for (int j = 0; j < 8; ++j) {
          float d2 = fmaxf(qn8[j] + kn - 2.f * ca.f[j], 0.f);
          float x  = SCALEF * (1.f - __expf(-d2));
          const int m_abs = m0 + j + 8 * hi;
          xs[j] = (s_abs > m_abs) ? (SCALEF * NEG_INF_C) : x;
        }
      };
      score_tile(0, x0);
      if (sub1) score_tile(16, x1);

      // ---------------- online softmax update ----------------
      float corr[8];
      #pragma unroll
      for (int j = 0; j < 8; ++j) {
        float xm = sub1 ? fmaxf(x0[j], x1[j]) : x0[j];
        xm = fmaxf(xm, __shfl_xor(xm, 1, 32));
        xm = fmaxf(xm, __shfl_xor(xm, 2, 32));
        xm = fmaxf(xm, __shfl_xor(xm, 4, 32));
        xm = fmaxf(xm, __shfl_xor(xm, 8, 32));
        const float mnew = fmaxf(m_i[j], xm);
        corr[j] = __expf(m_i[j] - mnew);
        const float p0 = __expf(x0[j] - mnew);
        const float p1 = sub1 ? __expf(x1[j] - mnew) : 0.f;
        float ps = p0 + p1;
        ps += __shfl_xor(ps, 1, 32);
        ps += __shfl_xor(ps, 2, 32);
        ps += __shfl_xor(ps, 4, 32);
        ps += __shfl_xor(ps, 8, 32);
        l_i[j] = l_i[j] * corr[j] + ps;
        m_i[j] = mnew;
        // C-layout -> LDS (relayout to A-layout for the A*V wmma)
        sP[wave][(j + 8*hi) * STILE + r16]      = (_Float16)p0;
        sP[wave][(j + 8*hi) * STILE + 16 + r16] = (_Float16)p1;
      }

      // P as A fragment (16x32 f16)
      U16H pf;
      const _Float16* pb = &sP[wave][r16 * STILE];
      pf.h8[0] = *(const v8h*)(pb + hi*8);
      pf.h8[1] = *(const v8h*)(pb + 16 + hi*8);

      // acc = acc*corr + P x V   (V from transposed LDS tile)
      #pragma unroll
      for (int cc = 0; cc < 4; ++cc) {
        const _Float16* vb = &sVT[(cc*16 + r16) * STILE];
        U16H vf;
        vf.h8[0] = *(const v8h*)(vb + hi*8);
        vf.h8[1] = *(const v8h*)(vb + 16 + hi*8);
        U8F a; a.v = acc[cc];
        #pragma unroll
        for (int j = 0; j < 8; ++j) a.f[j] *= corr[j];
        acc[cc] = __builtin_amdgcn_wmma_f32_16x16x32_f16(false, pf.v, false, vf.v,
                                                         (short)0, a.v, false, false);
      }
    }
    __syncthreads();
  }

  // ---------------- epilogue: normalize and store [B,L,H,D] ----------------
  float inv_l[8];
  #pragma unroll
  for (int j = 0; j < 8; ++j) inv_l[j] = 1.f / l_i[j];

  #pragma unroll
  for (int cc = 0; cc < 4; ++cc) {
    U8F a; a.v = acc[cc];
    #pragma unroll
    for (int j = 0; j < 8; ++j) {
      const int l_abs = m0 + j + 8 * hi;
      Out[(((size_t)(b * L_SZ + l_abs)) * H_SZ + h) * E_SZ + cc*16 + r16] =
          a.f[j] * inv_l[j];
    }
  }
}

extern "C" void kernel_launch(void* const* d_in, const int* in_sizes, int n_in,
                              void* d_out, int out_size, void* d_ws, size_t ws_size,
                              hipStream_t stream) {
  const float* Q = (const float*)d_in[0];
  const float* K = (const float*)d_in[1];
  const float* V = (const float*)d_in[2];
  float* Out = (float*)d_out;
  (void)in_sizes; (void)n_in; (void)out_size; (void)d_ws; (void)ws_size;

  dim3 grid(L_SZ / MTILE, 2 * H_SZ);   // (16 l-tiles, 16 batch*head problems)
  dim3 block(256);                     // 8 wave32 waves
  degradation_attn_fwd<<<grid, block, 0, stream>>>(Q, K, V, Out);
}
